// ContextEncoder_45191645888566
// MI455X (gfx1250) — compile-verified, matrix operand order checked by reading, not run on hardware
//
#include <hip/hip_runtime.h>
#include <hip/hip_bf16.h>
#include <cmath>

// ---------------------------------------------------------------------------
// Problem constants (from reference): B=16, T=128, D=64, NF=32, TS=64, H=128
// ---------------------------------------------------------------------------
constexpr int Bc  = 16;
constexpr int Tc  = 128;
constexpr int Dc  = 64;
constexpr int NFc = 32;
constexpr int TSc = 64;
constexpr int Hc  = 128;
constexpr int BP  = Bc * Dc;        // 1024 independent sequences
constexpr int Gc  = 4 * Hc;         // 512 gates per direction

typedef __attribute__((ext_vector_type(16))) _Float16 v16h;
typedef __attribute__((ext_vector_type(8)))  _Float16 v8h;
typedef __attribute__((ext_vector_type(4)))  _Float16 v4h;
typedef __attribute__((ext_vector_type(8)))  float    v8f;

// Workspace layout (bytes)
constexpr size_t OFF_WT16   = 0;                                   // 64x32 f16
constexpr size_t OFF_WIHF16 = OFF_WT16   + (size_t)TSc*NFc*2;      // 512x64 f16
constexpr size_t OFF_WHHF16 = OFF_WIHF16 + (size_t)Gc*TSc*2;       // 512x128 f16
constexpr size_t OFF_WIHB16 = OFF_WHHF16 + (size_t)Gc*Hc*2;
constexpr size_t OFF_WHHB16 = OFF_WIHB16 + (size_t)Gc*TSc*2;
constexpr size_t OFF_XPROJ  = OFF_WHHB16 + (size_t)Gc*Hc*2;        // [1024][128][64] f16
constexpr size_t OFF_HVEC   = OFF_XPROJ  + (size_t)BP*Tc*TSc*2;    // [1024][128][256] f16
constexpr size_t OFF_POOLED = OFF_HVEC   + (size_t)BP*Tc*2*Hc*2;   // [1024][256] f32
constexpr size_t WS_NEEDED  = OFF_POOLED + (size_t)BP*2*Hc*4;      // ~81.4 MiB

// ---------------------------------------------------------------------------
// Branch-free transcendental helpers (v_exp_f32 / v_rcp_f32, no EXEC games)
// ---------------------------------------------------------------------------
__device__ __forceinline__ float sig_fast(float x) {
  return __builtin_amdgcn_rcpf(1.0f + __expf(-x));
}
__device__ __forceinline__ float tanh_fast(float x) {
  const float ax = fabsf(x);
  const float e  = __expf(-2.0f * ax);
  const float t  = (1.0f - e) * __builtin_amdgcn_rcpf(1.0f + e);
  return copysignf(t, x);
}

// Load a 16-half WMMA operand fragment from a row-major [N][K] (or [M][K]) array.
// Per CDNA5 ISA layout: element e maps to k = kbase + 16*(e>>3) + 8*group + (e&7)
// i.e. two contiguous 8-half (16B) chunks -> two b128 loads.
__device__ __forceinline__ v16h load_frag(const _Float16* rowp, int kbase, int group) {
  const v8h* p = (const v8h*)(rowp + kbase + 8 * group);
  const v8h lo = p[0];
  const v8h hi = p[2];   // +16 halves
  v16h r;
#pragma unroll
  for (int e = 0; e < 8; ++e) {
    r[e]     = lo[e];
    r[8 + e] = hi[e];
  }
  return r;
}

__device__ __forceinline__ v8f wmma_f16(v16h a, v16h b, v8f c) {
  // (neg_a, A, neg_b, B, c_mod, C, reuse_a, reuse_b)
  return __builtin_amdgcn_wmma_f32_16x16x32_f16(false, a, false, b, (short)0, c, false, false);
}

// ---------------------------------------------------------------------------
// Kernel 0: fp32 -> fp16 weight conversion
// ---------------------------------------------------------------------------
__global__ void k0_convert(const float* __restrict__ src, _Float16* __restrict__ dst, int n) {
  int i = blockIdx.x * 256 + threadIdx.x;
  if (i < n) dst[i] = (_Float16)src[i];
}

// ---------------------------------------------------------------------------
// Kernel 1: xproj[r][ts] = tanh(X_row @ W_t^T + b_t), r = seq*T + t (131072 rows)
// One wave per 16-row tile; 4 WMMA (N-tiles) with K=32.
// ---------------------------------------------------------------------------
__global__ void __launch_bounds__(32) k1_xproj(const float* __restrict__ X,
                                               const _Float16* __restrict__ Wt16,
                                               const float* __restrict__ bt,
                                               _Float16* __restrict__ xproj) {
  const int lane  = threadIdx.x;
  const int group = lane >> 4;
  const int nlo   = lane & 15;

  const int r0  = blockIdx.x * 16;     // 16 rows, same sequence (T % 16 == 0)
  const int seq = r0 >> 7;             // r = seq*128 + t
  const int t0  = r0 & 127;
  const int bb  = seq >> 6;
  const int dd  = seq & 63;

  // A fragment: row M = lane&15, K=0..31 (fp32 source -> f16)
  const int t = t0 + nlo;
  const float* xr = X + (((size_t)bb * Tc + t) * Dc + dd) * NFc;
  v16h a;
#pragma unroll
  for (int e = 0; e < 8; ++e) {
    a[e]     = (_Float16)xr[8 * group + e];
    a[8 + e] = (_Float16)xr[16 + 8 * group + e];
  }

#pragma unroll
  for (int nt = 0; nt < 4; ++nt) {
    const int n = nt * 16 + nlo;
    v16h bf = load_frag(Wt16 + (size_t)n * NFc, 0, group);
    v8f acc = {};
    acc = wmma_f16(a, bf, acc);
    const float bias = bt[n];
#pragma unroll
    for (int v = 0; v < 8; ++v) {
      const int M = v + 8 * group;
      xproj[(size_t)(r0 + M) * TSc + nt * 16 + nlo] = (_Float16)tanh_fast(acc[v] + bias);
    }
  }
}

// ---------------------------------------------------------------------------
// Kernel 2: persistent bidirectional LSTM. Grid = 64 M-tiles x 2 directions.
// 8 waves/block; each wave owns 64 gate columns with Wih/Whh fragments held
// in VGPRs for all 128 time steps. c-state in registers, h via LDS tile.
// ---------------------------------------------------------------------------
__global__ void __launch_bounds__(256, 1) k2_lstm(const _Float16* __restrict__ xproj,
                                                  const _Float16* __restrict__ WihF,
                                                  const _Float16* __restrict__ WhhF,
                                                  const _Float16* __restrict__ WihB,
                                                  const _Float16* __restrict__ WhhB,
                                                  const float* __restrict__ bihF,
                                                  const float* __restrict__ bhhF,
                                                  const float* __restrict__ bihB,
                                                  const float* __restrict__ bhhB,
                                                  _Float16* __restrict__ hvec) {
  __shared__ __align__(16) _Float16 xS[16 * TSc];   // 2 KB  current x tile
  __shared__ __align__(16) _Float16 hS[16 * Hc];    // 4 KB  current h tile
  __shared__ __align__(16) float    gS[16 * Gc];    // 32 KB gate pre-activations

  const int dir     = blockIdx.x & 1;
  const int seqbase = (blockIdx.x >> 1) * 16;

  const _Float16* Wih = dir ? WihB : WihF;   // [512][64]
  const _Float16* Whh = dir ? WhhB : WhhF;   // [512][128]
  const float*    bih = dir ? bihB : bihF;
  const float*    bhh = dir ? bhhB : bhhF;

  const int tid   = threadIdx.x;
  const int wv    = tid >> 5;
  const int lane  = tid & 31;
  const int group = lane >> 4;
  const int nlo   = lane & 15;

  // ---- Weight fragments resident in VGPRs for the whole time loop ----
  v16h fih[4][2];   // x-projection weights, K=64 -> 2 k-steps
  v16h fhh[4][4];   // recurrent weights,    K=128 -> 4 k-steps
#pragma unroll
  for (int nt = 0; nt < 4; ++nt) {
    const int n = wv * 64 + nt * 16 + nlo;   // gate column (B row of W)
#pragma unroll
    for (int ks = 0; ks < 2; ++ks) fih[nt][ks] = load_frag(Wih + (size_t)n * TSc, 32 * ks, group);
#pragma unroll
    for (int ks = 0; ks < 4; ++ks) fhh[nt][ks] = load_frag(Whh + (size_t)n * Hc, 32 * ks, group);
  }

  // ---- Elementwise ownership: thread -> fixed row m, 8 consecutive units ----
  const int m  = tid >> 4;          // 0..15
  const int jb = (tid & 15) * 8;    // unit base 0..120
  float bsum[4][8];
  float cst[8];
#pragma unroll
  for (int gi = 0; gi < 4; ++gi)
#pragma unroll
    for (int i = 0; i < 8; ++i) bsum[gi][i] = bih[gi * Hc + jb + i] + bhh[gi * Hc + jb + i];
#pragma unroll
  for (int i = 0; i < 8; ++i) cst[i] = 0.0f;

  // zero initial h tile (16-byte stores)
  {
    v8h z = {};
    ((v8h*)hS)[tid] = z;
  }

  for (int t = 0; t < Tc; ++t) {
    __syncthreads();  // h/gate consumers from prev step done; hS init visible

    // stage x tile (16 seq x 64) from precomputed fp16 xproj; 8B per thread
    {
      const int mm  = tid >> 4;         // (tid*4)>>6
      const int col = (tid * 4) & 63;
      ((v4h*)xS)[tid] =
          *(const v4h*)&xproj[(((size_t)(seqbase + mm)) * Tc + t) * TSc + col];
    }
    __syncthreads();

    // A fragments (shared across this wave's 4 N-tiles)
    v16h ax0 = load_frag(xS + nlo * TSc, 0,  group);
    v16h ax1 = load_frag(xS + nlo * TSc, 32, group);
    v16h ah0 = load_frag(hS + nlo * Hc,  0,  group);
    v16h ah1 = load_frag(hS + nlo * Hc,  32, group);
    v16h ah2 = load_frag(hS + nlo * Hc,  64, group);
    v16h ah3 = load_frag(hS + nlo * Hc,  96, group);

#pragma unroll
    for (int nt = 0; nt < 4; ++nt) {
      v8f acc = {};
      acc = wmma_f16(ax0, fih[nt][0], acc);
      acc = wmma_f16(ax1, fih[nt][1], acc);
      acc = wmma_f16(ah0, fhh[nt][0], acc);
      acc = wmma_f16(ah1, fhh[nt][1], acc);
      acc = wmma_f16(ah2, fhh[nt][2], acc);
      acc = wmma_f16(ah3, fhh[nt][3], acc);
      const int n0 = wv * 64 + nt * 16;
#pragma unroll
      for (int v = 0; v < 8; ++v) gS[(v + 8 * group) * Gc + n0 + nlo] = acc[v];
    }
    __syncthreads();

    // LSTM cell nonlinearity; c in registers, h buffered then stored 16B-wide
    v8h hbuf;
#pragma unroll
    for (int i = 0; i < 8; ++i) {
      const int j = jb + i;
      const float ig = gS[m * Gc + j]            + bsum[0][i];
      const float fg = gS[m * Gc + Hc + j]       + bsum[1][i];
      const float gg = gS[m * Gc + 2 * Hc + j]   + bsum[2][i];
      const float og = gS[m * Gc + 3 * Hc + j]   + bsum[3][i];
      const float cc = sig_fast(fg) * cst[i] + sig_fast(ig) * tanh_fast(gg);
      const float hh = sig_fast(og) * tanh_fast(cc);
      cst[i]  = cc;
      hbuf[i] = (_Float16)hh;
    }
    *(v8h*)&hS[m * Hc + jb] = hbuf;
    *(v8h*)&hvec[(((size_t)(seqbase + m)) * Tc + t) * (2 * Hc) + dir * Hc + jb] = hbuf;
  }
}

// ---------------------------------------------------------------------------
// Kernel 3: attention pooling per sequence. scores = hvec . h_last, softmax_T,
// pooled = sum_t w_t * hvec[t].  One block (128 thr) per sequence.
// ---------------------------------------------------------------------------
__global__ void __launch_bounds__(128) k3_attn(const _Float16* __restrict__ hvec,
                                               float* __restrict__ pooled) {
  const int seq = blockIdx.x;
  const int tid = threadIdx.x;
  __shared__ _Float16 hLast[2 * Hc];
  __shared__ float    sc[Tc];
  __shared__ float    red[Tc];

  const _Float16* hs = hvec + (size_t)seq * Tc * (2 * Hc);
  hLast[tid]       = hs[(size_t)(Tc - 1) * 256 + tid];
  hLast[tid + 128] = hs[(size_t)(Tc - 1) * 256 + tid + 128];
  __syncthreads();

  float s = 0.0f;
  for (int ch = 0; ch < 256; ++ch) s += (float)hs[(size_t)tid * 256 + ch] * (float)hLast[ch];
  red[tid] = s;
  __syncthreads();
  for (int off = 64; off > 0; off >>= 1) {
    if (tid < off) red[tid] = fmaxf(red[tid], red[tid + off]);
    __syncthreads();
  }
  const float mx = red[0];
  __syncthreads();
  const float e = __expf(s - mx);
  red[tid] = e;
  __syncthreads();
  for (int off = 64; off > 0; off >>= 1) {
    if (tid < off) red[tid] += red[tid + off];
    __syncthreads();
  }
  const float inv = __builtin_amdgcn_rcpf(red[0]);
  __syncthreads();
  sc[tid] = e * inv;
  __syncthreads();

#pragma unroll
  for (int c0 = 0; c0 < 2; ++c0) {
    const int ch = tid + c0 * 128;
    float acc = 0.0f;
    for (int t = 0; t < Tc; ++t) acc += sc[t] * (float)hs[(size_t)t * 256 + ch];
    pooled[(size_t)seq * 256 + ch] = acc;
  }
}

// ---------------------------------------------------------------------------
// Kernel 4: per-batch-row normalization (mean/std ddof=1 over 16384) + affine.
// ---------------------------------------------------------------------------
__global__ void __launch_bounds__(256) k4_norm(const float* __restrict__ pooled,
                                               const float* __restrict__ dw,
                                               const float* __restrict__ db,
                                               float* __restrict__ out) {
  const int b   = blockIdx.x;
  const int tid = threadIdx.x;
  const int N   = Dc * 2 * Hc;  // 16384
  const float* row = pooled + (size_t)b * N;
  __shared__ float rs[256];
  __shared__ float rq[256];
  float s = 0.0f, q = 0.0f;
  for (int i = tid; i < N; i += 256) {
    const float v = row[i];
    s += v;
    q += v * v;
  }
  rs[tid] = s;
  rq[tid] = q;
  __syncthreads();
  for (int off = 128; off > 0; off >>= 1) {
    if (tid < off) {
      rs[tid] += rs[tid + off];
      rq[tid] += rq[tid + off];
    }
    __syncthreads();
  }
  const float mean = rs[0] / (float)N;
  const float var  = (rq[0] - (float)N * mean * mean) / (float)(N - 1);
  const float inv  = rsqrtf(var);
  for (int i = tid; i < N; i += 256)
    out[(size_t)b * N + i] = (row[i] - mean) * inv * dw[i] + db[i];
}

// ---------------------------------------------------------------------------
// Host launcher
// ---------------------------------------------------------------------------
extern "C" void kernel_launch(void* const* d_in, const int* in_sizes, int n_in,
                              void* d_out, int out_size, void* d_ws, size_t ws_size,
                              hipStream_t stream) {
  (void)in_sizes; (void)n_in; (void)out_size; (void)ws_size; // needs WS_NEEDED (~82 MiB)

  const float* X      = (const float*)d_in[0];
  const float* W_t    = (const float*)d_in[1];
  const float* b_t    = (const float*)d_in[2];
  const float* Wih_f  = (const float*)d_in[3];
  const float* Whh_f  = (const float*)d_in[4];
  const float* bih_f  = (const float*)d_in[5];
  const float* bhh_f  = (const float*)d_in[6];
  const float* Wih_b  = (const float*)d_in[7];
  const float* Whh_b  = (const float*)d_in[8];
  const float* bih_b  = (const float*)d_in[9];
  const float* bhh_b  = (const float*)d_in[10];
  const float* diag_w = (const float*)d_in[11];
  const float* diag_b = (const float*)d_in[12];

  char* ws = (char*)d_ws;
  _Float16* Wt16   = (_Float16*)(ws + OFF_WT16);
  _Float16* WihF16 = (_Float16*)(ws + OFF_WIHF16);
  _Float16* WhhF16 = (_Float16*)(ws + OFF_WHHF16);
  _Float16* WihB16 = (_Float16*)(ws + OFF_WIHB16);
  _Float16* WhhB16 = (_Float16*)(ws + OFF_WHHB16);
  _Float16* xproj  = (_Float16*)(ws + OFF_XPROJ);
  _Float16* hvec   = (_Float16*)(ws + OFF_HVEC);
  float*    pooled = (float*)(ws + OFF_POOLED);

  // Weight conversion to fp16
  k0_convert<<<(TSc * NFc + 255) / 256, 256, 0, stream>>>(W_t,   Wt16,   TSc * NFc);
  k0_convert<<<(Gc * TSc + 255) / 256, 256, 0, stream>>>(Wih_f, WihF16, Gc * TSc);
  k0_convert<<<(Gc * Hc + 255) / 256, 256, 0, stream>>>(Whh_f, WhhF16, Gc * Hc);
  k0_convert<<<(Gc * TSc + 255) / 256, 256, 0, stream>>>(Wih_b, WihB16, Gc * TSc);
  k0_convert<<<(Gc * Hc + 255) / 256, 256, 0, stream>>>(Whh_b, WhhB16, Gc * Hc);

  // tanh input projection (WMMA GEMM, 131072x64, K=32)
  k1_xproj<<<(BP * Tc) / 16, 32, 0, stream>>>(X, Wt16, b_t, xproj);

  // Recurrent dual LSTM (WMMA, weight-stationary in VGPRs)
  k2_lstm<<<(BP / 16) * 2, 256, 0, stream>>>(xproj, WihF16, WhhF16, WihB16, WhhB16,
                                             bih_f, bhh_f, bih_b, bhh_b, hvec);

  // Attention pooling
  k3_attn<<<BP, 128, 0, stream>>>(hvec, pooled);

  // Final normalization + affine
  k4_norm<<<Bc, 256, 0, stream>>>(pooled, diag_w, diag_b, (float*)d_out);
}
